// Grace_42992622633744
// MI455X (gfx1250) — matrix operand order repaired and checked by reading, not compiled, not run on hardware
//
#include <hip/hip_runtime.h>

#define N_NODES 100000
#define N_EDGES 1600000
// layer dims: IN=128 -> HID2=128 -> OUT=64; concat-K = 256 for both layers

typedef __attribute__((ext_vector_type(16))) __bf16 v16bf;
typedef __attribute__((ext_vector_type(8)))  float  v8f;

__device__ __forceinline__ unsigned int f2bf(float f) {
    unsigned int u = __float_as_uint(f);
    unsigned int lsb = (u >> 16) & 1u;
    u += 0x7fffu + lsb;                 // round-to-nearest-even
    return u >> 16;
}

__device__ __forceinline__ uint2 pack4bf(float4 v) {
    uint2 r;
    r.x = f2bf(v.x) | (f2bf(v.y) << 16);
    r.y = f2bf(v.z) | (f2bf(v.w) << 16);
    return r;
}

union FragB16 {
    v16bf v;
    uint4 q[2];
};

// ---------------- degree ----------------
__global__ void k_degree(const int* __restrict__ dst, float* __restrict__ degf) {
    int e = blockIdx.x * blockDim.x + threadIdx.x;
    if (e < N_EDGES) atomicAdd(&degf[dst[e]], 1.0f);
}

__global__ void k_invdeg(float* __restrict__ degf) {
    int i = blockIdx.x * blockDim.x + threadIdx.x;
    if (i < N_NODES) degf[i] = 1.0f / fmaxf(degf[i], 1.0f);
}

// ------- scatter: aggF[dst] += feat[src]; one wave32 per edge, 128 floats -------
// Node matrix (51.2 MB) is L2-resident on MI455X (192 MB L2): both the random
// gather and the fp32 atomic adds resolve on-chip.
__global__ void k_scatter128(const float* __restrict__ feat,
                             const int* __restrict__ src,
                             const int* __restrict__ dst,
                             float* __restrict__ aggF) {
    int e    = blockIdx.x * 8 + (threadIdx.x >> 5);
    int lane = threadIdx.x & 31;
    if (e >= N_EDGES) return;
    int s = src[e];
    int d = dst[e];
    const float4 v = *(const float4*)(feat + (size_t)s * 128 + lane * 4);
    float* o = aggF + (size_t)d * 128 + lane * 4;
    atomicAdd(o + 0, v.x);
    atomicAdd(o + 1, v.y);
    atomicAdd(o + 2, v.z);
    atomicAdd(o + 3, v.w);
}

// ---------------- fp32 -> bf16, vectorized 4-wide ----------------
__global__ void k_cvt4(const float4* __restrict__ in, uint2* __restrict__ out, int n4) {
    int i = blockIdx.x * blockDim.x + threadIdx.x;
    if (i < n4) out[i] = pack4bf(in[i]);
}

// ------- normalize accumulated sum by inverse degree + convert, 4-wide -------
__global__ void k_normcvt4(const float4* __restrict__ aggF,
                           const float* __restrict__ invdeg,
                           uint2* __restrict__ out) {
    int i = blockIdx.x * blockDim.x + threadIdx.x;
    if (i >= N_NODES * 32) return;               // 128/4 = 32 float4 per row
    float s = invdeg[i >> 5];
    float4 v = aggF[i];
    v.x *= s; v.y *= s; v.z *= s; v.w *= s;
    out[i] = pack4bf(v);
}

// ------- pack concat weights [Wself;Wneigh] (256 x H fp32) into bf16 B-fragments -------
// Bpack[(((ct*8)+t)*32 + l)*16 + j] = Wcat[ t*32 + (l>>4)*8 + (j&7) + (j>>3)*16 ][ ct*16 + (l&15) ]
// (lanes 0-15 hold K {0..7,16..23} of the 32-wide K-step, lanes 16-31 hold K {8..15,24..31}; n = lane&15)
__global__ void k_pack(const float* __restrict__ Wself, const float* __restrict__ Wneigh,
                       unsigned short* __restrict__ Bpack, int H) {
    int idx = blockIdx.x * blockDim.x + threadIdx.x;
    if (idx >= 256 * H) return;
    int j  = idx & 15;
    int l  = (idx >> 4) & 31;
    int t  = (idx >> 9) & 7;
    int ct = idx >> 12;
    int k  = t * 32 + (l >> 4) * 8 + (j & 7) + (j >> 3) * 16;
    int n  = ct * 16 + (l & 15);
    float w = (k < 128) ? Wself[k * H + n] : Wneigh[(k - 128) * H + n];
    Bpack[idx] = (unsigned short)f2bf(w);
}

// ------- fused SAGE GEMM: out = relu([selfB | aggB] @ Wcat + bias) -------
// one 16x16 output tile per wave; K=256 as 8 x v_wmma_f32_16x16x32_bf16.
// EMIT_BF16: also write the bf16 copy of the activation (feeds next layer's
// A-matrix), saving a full 77 MB convert pass.
template <int H, bool EMIT_BF16>
__global__ void __launch_bounds__(32 * (H / 16))
k_sage_gemm(const unsigned short* __restrict__ selfB,
            const unsigned short* __restrict__ aggB,
            const unsigned short* __restrict__ Bpack,
            const float* __restrict__ bias,
            float* __restrict__ out,
            unsigned short* __restrict__ outB) {
    const int lane    = threadIdx.x & 31;
    const int w       = threadIdx.x >> 5;      // column tile index
    const int rowbase = blockIdx.x * 16;       // grid = 6250 exactly (no divergence)
    const int half    = lane >> 4;
    const size_t mrow = (size_t)(rowbase + (lane & 15)) * 128;

    v8f c = {};
#pragma unroll
    for (int t = 0; t < 8; ++t) {
        // A fragment (16-bit A layout): lanes 0-15 -> K {0..7,16..23}+K0,
        //                               lanes 16-31 -> K {8..15,24..31}+K0
        const unsigned short* asrc = (t < 4) ? selfB : aggB;
        const int koff = (t & 3) * 32 + half * 8;
        FragB16 a, b;
        a.q[0] = *(const uint4*)(asrc + mrow + koff);
        a.q[1] = *(const uint4*)(asrc + mrow + koff + 16);
        // B fragment: 16 contiguous bf16 per lane from the packed buffer
        const unsigned short* bsrc = Bpack + (((size_t)w * 8 + t) * 32 + lane) * 16;
        b.q[0] = *(const uint4*)(bsrc);
        b.q[1] = *(const uint4*)(bsrc + 8);
        c = __builtin_amdgcn_wmma_f32_16x16x32_bf16(false, a.v, false, b.v,
                                                    (short)0, c, false, false);
    }

    // C/D layout: VGPR r, lanes 0-15 -> M=r, lanes 16-31 -> M=8+r; N = lane&15
    const int col = w * 16 + (lane & 15);
    const float bv = bias[col];
#pragma unroll
    for (int r = 0; r < 8; ++r) {
        const int row = rowbase + half * 8 + r;
        const float v = fmaxf(c[r] + bv, 0.0f);
        out[(size_t)row * H + col] = v;
        if (EMIT_BF16) outB[(size_t)row * H + col] = (unsigned short)f2bf(v);
    }
}

extern "C" void kernel_launch(void* const* d_in, const int* in_sizes, int n_in,
                              void* d_out, int out_size, void* d_ws, size_t ws_size,
                              hipStream_t stream) {
    const float* h    = (const float*)d_in[0];
    const int*   esrc = (const int*)  d_in[1];
    const int*   edst = (const int*)  d_in[2];
    const float* Ws1  = (const float*)d_in[3];
    const float* Wn1  = (const float*)d_in[4];
    const float* b1   = (const float*)d_in[5];
    const float* Ws2  = (const float*)d_in[6];
    const float* Wn2  = (const float*)d_in[7];
    const float* b2   = (const float*)d_in[8];
    float* out = (float*)d_out;

    // ---- workspace carve-up (256B aligned) ----
    char* ws = (char*)d_ws;
    size_t o = 0;
    auto take = [&](size_t bytes) -> void* {
        void* p = ws + o;
        o += (bytes + 255) & ~(size_t)255;
        return p;
    };
    const size_t NF = (size_t)N_NODES * 128;
    float*          invdeg = (float*)         take((size_t)N_NODES * 4);
    float*          aggF   = (float*)         take(NF * 4);   // fp32 scatter accumulator
    unsigned short* selfB  = (unsigned short*)take(NF * 2);   // bf16 self features
    unsigned short* aggB   = (unsigned short*)take(NF * 2);   // bf16 normalized agg
    float*          h1     = (float*)         take(NF * 4);   // layer-1 fp32 output
    unsigned short* bp1    = (unsigned short*)take((size_t)256 * 128 * 2);
    unsigned short* bp2    = (unsigned short*)take((size_t)256 * 64 * 2);
    (void)ws_size; (void)in_sizes; (void)n_in; (void)out_size;

    const int TB = 256;
    const int gE   = (N_EDGES + TB - 1) / TB;        // 6250
    const int gN   = (N_NODES + TB - 1) / TB;        // 391
    const int gNF4 = (int)((NF / 4 + TB - 1) / TB);  // 12500 (4-wide passes)
    const int gSC  = (N_EDGES + 7) / 8;              // 200000 (8 edges / block)
    const int gMM  = N_NODES / 16;                   // 6250 (exact)

    // ---- graph-invariant prep ----
    hipMemsetAsync(invdeg, 0, (size_t)N_NODES * 4, stream);
    hipMemsetAsync(aggF,   0, NF * 4, stream);
    k_pack<<<(256 * 128 + TB - 1) / TB, TB, 0, stream>>>(Ws1, Wn1, bp1, 128);
    k_pack<<<(256 * 64  + TB - 1) / TB, TB, 0, stream>>>(Ws2, Wn2, bp2, 64);
    k_degree<<<gE, TB, 0, stream>>>(edst, invdeg);
    k_invdeg<<<gN, TB, 0, stream>>>(invdeg);

    // ---- layer 1 ----
    k_cvt4<<<gNF4, TB, 0, stream>>>((const float4*)h, (uint2*)selfB, (int)(NF / 4));
    k_scatter128<<<gSC, TB, 0, stream>>>(h, esrc, edst, aggF);
    k_normcvt4<<<gNF4, TB, 0, stream>>>((const float4*)aggF, invdeg, (uint2*)aggB);
    // writes fp32 h1 (for the layer-2 scatter gather) AND bf16 selfB (layer-2 A-matrix)
    k_sage_gemm<128, true><<<gMM, 256, 0, stream>>>(selfB, aggB, bp1, b1, h1, selfB);

    // ---- layer 2 ----
    hipMemsetAsync(aggF, 0, NF * 4, stream);
    k_scatter128<<<gSC, TB, 0, stream>>>(h1, esrc, edst, aggF);
    k_normcvt4<<<gNF4, TB, 0, stream>>>((const float4*)aggF, invdeg, (uint2*)aggB);
    k_sage_gemm<64, false><<<gMM, 128, 0, stream>>>(selfB, aggB, bp2, b2, out, nullptr);
}
